// MolVisGNN_2259152798258
// MI455X (gfx1250) — matrix operand order repaired and checked by reading, not compiled, not run on hardware
//
#include <hip/hip_runtime.h>
#include <cstdint>
#include <cstddef>

#define NN 4096
#define NE 131072
#define BN_INV 0.9999950000374997f   // 1/sqrt(1+1e-5)

typedef __attribute__((ext_vector_type(16))) _Float16 v16h;
typedef __attribute__((ext_vector_type(8)))  _Float16 v8h;
typedef __attribute__((ext_vector_type(8)))  float    v8f;

// ---- WMMA fragment helpers (wave32, 16x16x32 f16, f32 accum) ----------------
// A (16x32) per-lane layout: lane L holds row m=L%16; halves 0..15 = K[16*hi .. 16*hi+15]
// (hi = L/16)  -> two 16B loads.
__device__ __forceinline__ v16h ld_a_frag(const _Float16* p) {
  v8h lo = *(const v8h*)p;
  v8h hi = *(const v8h*)(p + 8);
  return __builtin_shufflevector(lo, hi, 0,1,2,3,4,5,6,7,8,9,10,11,12,13,14,15);
}

// ---- generic GEMM: C[M x Nc] = A[M x K](f16, row-major) * B^T --------------
// B given as Wh[Nc x K] row-major f16 (i.e. B[k][n] = Wh[n*K+k]); B layout used here:
// lane L holds col n=L%16, 16 contiguous K halves starting at 16*(L/16) -> one 32B load.
__global__ void __launch_bounds__(256) k_gemm(const _Float16* __restrict__ A,
                                              const _Float16* __restrict__ Wh,
                                              float* __restrict__ C,
                                              int M, int K, int Nc) {
  int wave = (blockIdx.x * 256 + threadIdx.x) >> 5;
  int lane = threadIdx.x & 31;
  int numNT = Nc >> 4;
  int nt = wave % numNT;
  int mb = (wave / numNT) * 64;
  int ln = lane & 15, hi = lane >> 4;
  const _Float16* bptr = Wh + (size_t)(nt * 16 + ln) * K + hi * 16;
  const _Float16* aptr = A  + (size_t)(mb + ln) * K + hi * 16;
  size_t rstep = (size_t)16 * K;
  v8f acc0 = {}, acc1 = {}, acc2 = {}, acc3 = {};
  for (int k0 = 0; k0 < K; k0 += 32) {
    v16h b = *(const v16h*)(bptr + k0);
    const _Float16* pa = aptr + k0;
    v16h a0 = ld_a_frag(pa);
    v16h a1 = ld_a_frag(pa + rstep);
    v16h a2 = ld_a_frag(pa + 2 * rstep);
    v16h a3 = ld_a_frag(pa + 3 * rstep);
    acc0 = __builtin_amdgcn_wmma_f32_16x16x32_f16(false, a0, false, b, (short)0, acc0, false, false);
    acc1 = __builtin_amdgcn_wmma_f32_16x16x32_f16(false, a1, false, b, (short)0, acc1, false, false);
    acc2 = __builtin_amdgcn_wmma_f32_16x16x32_f16(false, a2, false, b, (short)0, acc2, false, false);
    acc3 = __builtin_amdgcn_wmma_f32_16x16x32_f16(false, a3, false, b, (short)0, acc3, false, false);
  }
  // D layout: VGPR r, lane L -> m = r + 8*(L/16), n = L%16
  float* cb = C + (size_t)(mb + hi * 8) * Nc + nt * 16 + ln;
#pragma unroll
  for (int r = 0; r < 8; ++r) {
    cb[(size_t)(r)      * Nc] = acc0[r];
    cb[(size_t)(16 + r) * Nc] = acc1[r];
    cb[(size_t)(32 + r) * Nc] = acc2[r];
    cb[(size_t)(48 + r) * Nc] = acc3[r];
  }
}

// ---- misc elementwise kernels ----------------------------------------------
__global__ void k_zero(float* p, size_t n) {
  size_t i = (size_t)blockIdx.x * blockDim.x + threadIdx.x;
  if (i < n) p[i] = 0.f;
}

// f1 = avgpool4(drug_1d), f2 = avgpool8(drug_2d): N*256 threads
__global__ void k_pool12(const float* __restrict__ d1, const float* __restrict__ d2,
                         float* __restrict__ f1, float* __restrict__ f2) {
  int i = blockIdx.x * blockDim.x + threadIdx.x;
  int n = i >> 8, j = i & 255;
  const float* p1 = d1 + (size_t)n * 1024 + 4 * j;
  f1[i] = (p1[0] + p1[1] + p1[2] + p1[3]) * 0.25f;
  const float* p2 = d2 + (size_t)n * 2048 + 8 * j;
  float s = 0.f;
#pragma unroll
  for (int q = 0; q < 8; ++q) s += p2[q];
  f2[i] = s * 0.125f;
}

// Build effective conv weight Wh[c=o*64+oc][k=ic*8+v] (f16), 512*1024 threads
__global__ void k_conv_w(const float* __restrict__ w, _Float16* __restrict__ Wh) {
  int i = blockIdx.x * blockDim.x + threadIdx.x;
  int c = i >> 10, k = i & 1023;
  int o = c >> 6, oc = c & 63;
  int ic = k >> 3, v = k & 7;
  int od = (o >> 2) & 1, oh = (o >> 1) & 1, ow = o & 1;
  int vd = (v >> 2) & 1, vh = (v >> 1) & 1, vw = v & 1;
  int kd = vd - od + 1, kh = vh - oh + 1, kw = vw - ow + 1;
  Wh[(size_t)c * 1024 + k] = (_Float16)w[oc * 3456 + ic * 27 + kd * 9 + kh * 3 + kw];
}

// Build A (f16) for conv branch b from drug_3d: N*1024 threads
__global__ void k_a3d(const float* __restrict__ x, _Float16* __restrict__ A, int b) {
  size_t i = (size_t)blockIdx.x * blockDim.x + threadIdx.x;
  size_t n = i >> 10;
  int k = (int)(i & 1023);
  int ic = k >> 3, r = k & 7;
  A[i] = (_Float16)x[n * 3072 + ic * 24 + 8 * b + r];
}

// enc3sum[n][oc] (+)= sum_o relu(C[n][o*64+oc] + bias[oc]) : N*64 threads
__global__ void k_conv_reduce(const float* __restrict__ C, const float* __restrict__ bias,
                              float* __restrict__ e3, int accum) {
  int i = blockIdx.x * blockDim.x + threadIdx.x;
  int n = i >> 6, oc = i & 63;
  const float* c = C + (size_t)n * 512;
  float bb = bias[oc], s = 0.f;
#pragma unroll
  for (int o = 0; o < 8; ++o) s += fmaxf(c[o * 64 + oc] + bb, 0.f);
  if (accum) e3[i] += s; else e3[i] = s;
}

// Collapse linear gating MLP: v[256], c scalar. One block of 256 threads.
__global__ void k_branchvec(const float* g1w, const float* g1b, const float* g2w,
                            const float* g2b, const float* g3w, const float* g3b,
                            float* out) {
  __shared__ float t3[64], t2[128];
  int t = threadIdx.x;
  if (t < 64) { float s = 0.f; for (int j = 0; j < 32; ++j) s += g3w[j * 64 + t]; t3[t] = s * (1.f / 32.f); }
  __syncthreads();
  if (t < 128) { float s = 0.f; for (int k = 0; k < 64; ++k) s += g2w[k * 128 + t] * t3[k]; t2[t] = s; }
  __syncthreads();
  { float s = 0.f; for (int i2 = 0; i2 < 128; ++i2) s += g1w[i2 * 256 + t] * t2[i2]; out[t] = s; }
  if (t == 0) {
    float c = 0.f;
    for (int i2 = 0; i2 < 128; ++i2) c += g1b[i2] * t2[i2];
    for (int k = 0; k < 64; ++k) c += g2b[k] * t3[k];
    float s3 = 0.f; for (int j = 0; j < 32; ++j) s3 += g3b[j];
    out[256] = c + s3 * (1.f / 32.f);
  }
}

// gates: dots, sigmoid, softmax; write w to output; build weighted A768 (f16)
__global__ void k_gate(const float* __restrict__ f1, const float* __restrict__ f2,
                       const float* __restrict__ e3s, const float* __restrict__ vbr,
                       _Float16* __restrict__ A768, float* __restrict__ wout) {
  __shared__ float r1[256], r2[256], r3[256], wsh[3];
  int n = blockIdx.x, t = threadIdx.x;
  float v  = vbr[t];
  float x1 = f1[(size_t)n * 256 + t];
  float x2 = f2[(size_t)n * 256 + t];
  float x3 = e3s[(size_t)n * 64 + (t >> 2)] * (1.f / 24.f);
  r1[t] = x1 * v; r2[t] = x2 * v; r3[t] = x3 * v;
  __syncthreads();
  for (int s = 128; s > 0; s >>= 1) {
    if (t < s) { r1[t] += r1[t + s]; r2[t] += r2[t + s]; r3[t] += r3[t + s]; }
    __syncthreads();
  }
  if (t == 0) {
    float c = vbr[256];
    float d0 = 1.f / (1.f + expf(-(r1[0] + c)));
    float d1 = 1.f / (1.f + expf(-(r2[0] + c)));
    float d2 = 1.f / (1.f + expf(-(r3[0] + c)));
    float m = fmaxf(d0, fmaxf(d1, d2));
    float e0 = expf(d0 - m), e1 = expf(d1 - m), e2 = expf(d2 - m);
    float inv = 1.f / (e0 + e1 + e2);
    wsh[0] = e0 * inv; wsh[1] = e1 * inv; wsh[2] = e2 * inv;
    wout[n * 3 + 0] = wsh[0]; wout[n * 3 + 1] = wsh[1]; wout[n * 3 + 2] = wsh[2];
  }
  __syncthreads();
  _Float16* row = A768 + (size_t)n * 768;
  row[t]        = (_Float16)(x1 * wsh[1]);
  row[256 + t]  = (_Float16)(x2 * wsh[0]);
  row[512 + t]  = (_Float16)(x3 * wsh[2]);
}

__global__ void k_w2h(const float* __restrict__ in, _Float16* __restrict__ out, int n) {
  int i = blockIdx.x * blockDim.x + threadIdx.x;
  if (i < n) out[i] = (_Float16)in[i];
}

// feat = bn(C + lne_b): N*128 threads
__global__ void k_bnfeat(const float* __restrict__ C, const float* __restrict__ lb,
                         const float* __restrict__ g, const float* __restrict__ b,
                         float* __restrict__ feat) {
  int i = blockIdx.x * blockDim.x + threadIdx.x;
  int c = i & 127;
  feat[i] = (C[i] + lb[c]) * (g[c] * BN_INV) + b[c];
}

// ConvT1d tower: one 128-thread block per sample; feat += scalar res (in place)
__global__ void __launch_bounds__(128) k_tower(float* __restrict__ feat,
    const float* __restrict__ w1, const float* __restrict__ b1,
    const float* __restrict__ w2, const float* __restrict__ b2,
    const float* __restrict__ w3, const float* __restrict__ b3) {
  __shared__ float sx[128];
  __shared__ float z1[32 * 31];
  __shared__ float z2[64 * 7];
  __shared__ float red[128];
  int n = blockIdx.x, t = threadIdx.x;
  sx[t] = feat[(size_t)n * 128 + t];
  __syncthreads();
  for (int i = t; i < 32 * 31; i += 128) {
    int ch = i / 31, u = i % 31;
    float a0 = w1[ch * 2 + 0], a1 = w1[ch * 2 + 1], bb = b1[ch];
    float m = -3.4e38f;
#pragma unroll
    for (int q = 0; q < 4; ++q) {
      int tt = 4 * u + q;
      float y = a1 * sx[tt] + a0 * sx[tt + 1] + bb;
      y = y > 0.f ? y : 0.f;
      m = fmaxf(m, y);
    }
    z1[i] = m;
  }
  __syncthreads();
  for (int i = t; i < 64 * 7; i += 128) {
    int oc = i / 7, u = i % 7;
    float bb = b2[oc], m = -3.4e38f;
#pragma unroll
    for (int q = 0; q < 4; ++q) {
      int tt = 4 * u + q;
      float acc = bb;
      for (int ic = 0; ic < 32; ++ic)
        acc += w2[ic * 128 + oc * 2 + 1] * z1[ic * 31 + tt] + w2[ic * 128 + oc * 2] * z1[ic * 31 + tt + 1];
      acc = acc > 0.f ? acc : 0.f;
      m = fmaxf(m, acc);
    }
    z2[i] = m;
  }
  __syncthreads();
  {
    int oc = t;
    float bb = b3[oc], m = -3.4e38f;
#pragma unroll
    for (int q = 0; q < 4; ++q) {
      float acc = bb;
      for (int ic = 0; ic < 64; ++ic)
        acc += w3[ic * 256 + oc * 2 + 1] * z2[ic * 7 + q] + w3[ic * 256 + oc * 2] * z2[ic * 7 + q + 1];
      acc = acc > 0.f ? acc : 0.f;
      m = fmaxf(m, acc);
    }
    red[t] = m;
  }
  __syncthreads();
  for (int s = 64; s > 0; s >>= 1) { if (t < s) red[t] += red[t + s]; __syncthreads(); }
  feat[(size_t)n * 128 + t] = sx[t] + red[0] * (1.f / 128.f);
}

__global__ void k_deg(const int* __restrict__ dst, float* __restrict__ deg) {
  int e = blockIdx.x * blockDim.x + threadIdx.x;
  atomicAdd(&deg[dst[e]], 1.0f);
}

// scatter-add x[src] into agg[dst]; E*(D/4) threads
__global__ void k_scatter(const int* __restrict__ src, const int* __restrict__ dst,
                          const float* __restrict__ x, float* __restrict__ agg, int D) {
  size_t tid = (size_t)blockIdx.x * blockDim.x + threadIdx.x;
  int dq = D >> 2;
  int e = (int)(tid / dq);
  int c = (int)(tid % dq) * 4;
  const float4 v = *(const float4*)(x + (size_t)src[e] * D + c);
  float* p = agg + (size_t)dst[e] * D + c;
  atomicAdd(p + 0, v.x); atomicAdd(p + 1, v.y);
  atomicAdd(p + 2, v.z); atomicAdd(p + 3, v.w);
}

// A = f16([agg/deg , x]): N*2D threads
__global__ void k_concat(const float* __restrict__ agg, const float* __restrict__ deg,
                         const float* __restrict__ x, _Float16* __restrict__ A, int D) {
  size_t i = (size_t)blockIdx.x * blockDim.x + threadIdx.x;
  int D2 = 2 * D;
  size_t n = i / D2;
  int c = (int)(i % D2);
  float val = (c < D) ? agg[n * D + c] / fmaxf(deg[n], 1.f) : x[n * D + (c - D)];
  A[i] = (_Float16)val;
}

// Wh[c][k] = k<D ? wl[c][k] : wr[c][k-D]  (f16): Dout*2D threads
__global__ void k_wcat(const float* __restrict__ wl, const float* __restrict__ wr,
                       _Float16* __restrict__ Wh, int D, int Dout) {
  int i = blockIdx.x * blockDim.x + threadIdx.x;
  int D2 = 2 * D;
  int c = i / D2, k = i % D2;
  Wh[i] = (_Float16)((k < D) ? wl[c * D + k] : wr[c * D + (k - D)]);
}

// out = leaky_relu(bn(C + bl), 0.3): N*D threads
__global__ void k_sagepost(const float* __restrict__ C, const float* __restrict__ bl,
                           const float* __restrict__ g, const float* __restrict__ b,
                           float* __restrict__ out, int D) {
  size_t i = (size_t)blockIdx.x * blockDim.x + threadIdx.x;
  int c = (int)(i % D);
  float y = (C[i] + bl[c]) * (g[c] * BN_INV) + b[c];
  out[i] = y > 0.f ? y : 0.3f * y;
}

extern "C" void kernel_launch(void* const* d_in, const int* in_sizes, int n_in,
                              void* d_out, int out_size, void* d_ws, size_t ws_size,
                              hipStream_t stream) {
  (void)in_sizes; (void)n_in; (void)out_size; (void)ws_size;
  const float* drug1 = (const float*)d_in[0];
  const float* drug2 = (const float*)d_in[1];
  const float* drug3 = (const float*)d_in[2];
  const int*   eidx  = (const int*)d_in[3];
  const float* w3d[3] = {(const float*)d_in[4], (const float*)d_in[6], (const float*)d_in[8]};
  const float* b3d[3] = {(const float*)d_in[5], (const float*)d_in[7], (const float*)d_in[9]};
  const float* lne_w   = (const float*)d_in[10];
  const float* lne_b   = (const float*)d_in[11];
  const float* bng[3]  = {(const float*)d_in[12], (const float*)d_in[14], (const float*)d_in[16]};
  const float* bnb[3]  = {(const float*)d_in[13], (const float*)d_in[15], (const float*)d_in[17]};
  const float* ct1_w = (const float*)d_in[18]; const float* ct1_b = (const float*)d_in[19];
  const float* ct2_w = (const float*)d_in[20]; const float* ct2_b = (const float*)d_in[21];
  const float* ct3_w = (const float*)d_in[22]; const float* ct3_b = (const float*)d_in[23];
  const float* g1_w = (const float*)d_in[24]; const float* g1_b = (const float*)d_in[25];
  const float* g2_w = (const float*)d_in[26]; const float* g2_b = (const float*)d_in[27];
  const float* g3_w = (const float*)d_in[28]; const float* g3_b = (const float*)d_in[29];
  const float* s_wl[3] = {(const float*)d_in[30], (const float*)d_in[33], (const float*)d_in[36]};
  const float* s_bl[3] = {(const float*)d_in[31], (const float*)d_in[34], (const float*)d_in[37]};
  const float* s_wr[3] = {(const float*)d_in[32], (const float*)d_in[35], (const float*)d_in[38]};

  // workspace carve (256B aligned)
  char* base = (char*)d_ws;
  size_t off = 0;
  auto carve = [&](size_t bytes) -> char* {
    char* p = base + off;
    off = (off + bytes + 255) & ~(size_t)255;
    return p;
  };
  float*    f1     = (float*)   carve((size_t)NN * 256 * 4);
  float*    f2     = (float*)   carve((size_t)NN * 256 * 4);
  float*    e3s    = (float*)   carve((size_t)NN * 64 * 4);
  _Float16* A16    = (_Float16*)carve((size_t)NN * 1024 * 2);
  _Float16* Wh     = (_Float16*)carve((size_t)512 * 1024 * 2);
  float*    Cbuf   = (float*)   carve((size_t)NN * 512 * 4);
  float*    vbr    = (float*)   carve(257 * 4);
  float*    feat   = (float*)   carve((size_t)NN * 128 * 4);
  float*    deg    = (float*)   carve((size_t)NN * 4);
  float*    agg    = (float*)   carve((size_t)NN * 128 * 4);
  float*    h1     = (float*)   carve((size_t)NN * 128 * 4);
  float*    h2     = (float*)   carve((size_t)NN * 64 * 4);

  float* hout = (float*)d_out;            // h: N x 32
  float* wout = (float*)d_out + NN * 32;  // w: N x 3

  auto gemm = [&](const _Float16* A, const _Float16* B, float* C, int M, int K, int Nc) {
    int waves = (M / 64) * (Nc / 16);
    k_gemm<<<waves / 8, 256, 0, stream>>>(A, B, C, M, K, Nc);
  };

  // 1) pools
  k_pool12<<<NN * 256 / 256, 256, 0, stream>>>(drug1, drug2, f1, f2);

  // 2) conv3d branches as WMMA GEMMs -> relu-mean into e3s
  for (int b = 0; b < 3; ++b) {
    k_conv_w<<<512 * 1024 / 256, 256, 0, stream>>>(w3d[b], Wh);
    k_a3d<<<NN * 1024 / 256, 256, 0, stream>>>(drug3, A16, b);
    gemm(A16, Wh, Cbuf, NN, 1024, 512);
    k_conv_reduce<<<NN * 64 / 256, 256, 0, stream>>>(Cbuf, b3d[b], e3s, b == 0 ? 0 : 1);
  }

  // 3) collapsed gating MLP + softmax + weighted 768-feature assembly (f16)
  k_branchvec<<<1, 256, 0, stream>>>(g1_w, g1_b, g2_w, g2_b, g3_w, g3_b, vbr);
  k_gate<<<NN, 256, 0, stream>>>(f1, f2, e3s, vbr, A16, wout);

  // 4) lne GEMM (4096x768x128) + BN
  k_w2h<<<(128 * 768 + 255) / 256, 256, 0, stream>>>(lne_w, Wh, 128 * 768);
  gemm(A16, Wh, Cbuf, NN, 768, 128);
  k_bnfeat<<<NN * 128 / 256, 256, 0, stream>>>(Cbuf, lne_b, bng[0], bnb[0], feat);

  // 5) convT tower -> feat += res (in place)
  k_tower<<<NN, 128, 0, stream>>>(feat, ct1_w, ct1_b, ct2_w, ct2_b, ct3_w, ct3_b);

  // 6) degrees
  k_zero<<<(NN + 255) / 256, 256, 0, stream>>>(deg, NN);
  k_deg<<<NE / 256, 256, 0, stream>>>(eidx + NE, deg);

  // 7) three SAGE layers: [agg,x] @ [wl;wr]^T via WMMA
  int din[3]  = {128, 128, 64};
  int dout[3] = {128, 64, 32};
  const float* xin[3] = {feat, h1, h2};
  float* xout[3] = {h1, h2, hout};
  for (int l = 0; l < 3; ++l) {
    int D = din[l], Do = dout[l];
    k_zero<<<((size_t)NN * D + 255) / 256, 256, 0, stream>>>(agg, (size_t)NN * D);
    k_scatter<<<(int)((size_t)NE * (D / 4) / 256), 256, 0, stream>>>(eidx, eidx + NE, xin[l], agg, D);
    k_concat<<<(int)((size_t)NN * 2 * D / 256), 256, 0, stream>>>(agg, deg, xin[l], A16, D);
    k_wcat<<<(Do * 2 * D + 255) / 256, 256, 0, stream>>>(s_wl[l], s_wr[l], Wh, D, Do);
    gemm(A16, Wh, Cbuf, NN, 2 * D, Do);
    k_sagepost<<<(int)((size_t)NN * Do / 256), 256, 0, stream>>>(Cbuf, s_bl[l], bng[l], bnb[l], xout[l], Do);
  }
}
